// KMeansQuantizer_52922587021810
// MI455X (gfx1250) — matrix operand order repaired and checked by reading, not compiled, not run on hardware
//
#include <hip/hip_runtime.h>

// ---------------------------------------------------------------------------
// KMeans assignment on MI455X (gfx1250):
//   labels[n]   = argmin_k ( ||c_k||^2 - 2 * <x_n, c_k> )   (x_norm drops out)
//   assigned[n] = centroids[labels[n]]
// Core: v_wmma_f32_16x16x32_bf16 with bf16x3 split (hi*hi + hi*lo + lo*hi).
// Centroids pre-split to bf16 hi/lo once (prep kernel); B tiles staged to LDS
// via global_load_async_to_lds_b128 (ASYNCcnt), A tiles converted on the fly.
// ---------------------------------------------------------------------------

typedef __attribute__((ext_vector_type(16))) __bf16         v16bf;
typedef __attribute__((ext_vector_type(8)))  float          v8f;
typedef __attribute__((ext_vector_type(8)))  unsigned short v8us;
typedef __attribute__((ext_vector_type(16))) unsigned short v16us;
typedef __attribute__((ext_vector_type(4)))  int            v4i;

#define DDIM 1024          // feature dim
#define KCNT 1024          // number of centroids
#define LSTR 40            // LDS row stride in bf16 elems (80B: 16B-aligned, 64-bank conflict-free)
#define MTILE 128          // rows per block
#define NTILE 128          // centroid columns per jb block

// ---- async global->LDS copy (CDNA5), with safe fallback --------------------
#if __has_builtin(__builtin_amdgcn_global_load_async_to_lds_b128)
#define HAVE_ASYNC_LDS 1
__device__ __forceinline__ void async_copy_b128(const unsigned short* g, unsigned short* l) {
    // signature (from hipcc diagnostic): (AS1 int4*, AS3 int4*, imm offset, imm cpol)
    __builtin_amdgcn_global_load_async_to_lds_b128(
        (__attribute__((address_space(1))) v4i*)(g),
        (__attribute__((address_space(3))) v4i*)(l), 0, 0);
}
__device__ __forceinline__ void async_wait0() {
#if __has_builtin(__builtin_amdgcn_s_wait_asynccnt)
    __builtin_amdgcn_s_wait_asynccnt(0);
#else
    asm volatile("s_wait_asynccnt 0x0" ::: "memory");
#endif
}
#else
#define HAVE_ASYNC_LDS 0
__device__ __forceinline__ void async_copy_b128(const unsigned short* g, unsigned short* l) {
    *(v8us*)l = *(const v8us*)g;    // synchronous fallback (DScnt, handled by barrier)
}
__device__ __forceinline__ void async_wait0() {}
#endif

// ---- f32 <-> bf16 hi/lo split ----------------------------------------------
__device__ __forceinline__ unsigned short f2bf(float f) {
    unsigned u = __builtin_bit_cast(unsigned, f);
    u += 0x7FFFu + ((u >> 16) & 1u);      // round-to-nearest-even to bf16
    return (unsigned short)(u >> 16);
}
__device__ __forceinline__ float bf2f(unsigned short h) {
    unsigned u = ((unsigned)h) << 16;
    return __builtin_bit_cast(float, u);
}
__device__ __forceinline__ void cvt_store4(unsigned short* hi, unsigned short* lo, float4 v) {
    float f[4] = {v.x, v.y, v.z, v.w};
    union { unsigned short s[4]; uint2 u; } ph, pl;
#pragma unroll
    for (int i = 0; i < 4; ++i) {
        unsigned short h = f2bf(f[i]);
        ph.s[i] = h;
        pl.s[i] = f2bf(f[i] - bf2f(h));
    }
    *(uint2*)hi = ph.u;
    *(uint2*)lo = pl.u;
}

// ---- WMMA fragment loads from LDS (layouts per CDNA5 ISA 7.12.2) -----------
__device__ __forceinline__ v16bf make_frag(v8us p0, v8us p1) {
    v16us w = __builtin_shufflevector(p0, p1, 0,1,2,3,4,5,6,7,8,9,10,11,12,13,14,15);
    union { v16us u; v16bf b; } c; c.u = w;
    return c.b;
}
// A 16x32: half 0 -> K {0..7,16..23}, half 1 -> K {8..15,24..31}
__device__ __forceinline__ v16bf load_fragA(const unsigned short* row, int half) {
    v8us p0 = *(const v8us*)(row + half * 8);
    v8us p1 = *(const v8us*)(row + 16 + half * 8);
    return make_frag(p0, p1);
}
// B 32x16 (column = centroid row): half 0 -> K {0..7,16..23}, half 1 -> K {8..15,24..31}
__device__ __forceinline__ v16bf load_fragB(const unsigned short* row, int half) {
    v8us p0 = *(const v8us*)(row + half * 8);
    v8us p1 = *(const v8us*)(row + 16 + half * 8);
    return make_frag(p0, p1);
}

// ---------------------------------------------------------------------------
// Kernel 1 (prep): cnorm[k] = ||c_k||^2 ; split centroids into bf16 hi/lo.
// ---------------------------------------------------------------------------
__global__ __launch_bounds__(256) void kq_prep_kernel(const float* __restrict__ cents,
                                                      float* __restrict__ cnorm,
                                                      unsigned short* __restrict__ Chi,
                                                      unsigned short* __restrict__ Clo) {
    __shared__ float partial[8];
    const int k = blockIdx.x;
    const size_t base = (size_t)k * DDIM + threadIdx.x * 4;
    const float4 v = *(const float4*)&cents[base];
    cvt_store4(&Chi[base], &Clo[base], v);
    float s = v.x * v.x + v.y * v.y + v.z * v.z + v.w * v.w;
#pragma unroll
    for (int off = 16; off >= 1; off >>= 1) s += __shfl_down(s, off, 32);
    if ((threadIdx.x & 31) == 0) partial[threadIdx.x >> 5] = s;
    __syncthreads();
    if (threadIdx.x == 0) {
        float t = 0.f;
#pragma unroll
        for (int w = 0; w < 8; ++w) t += partial[w];
        cnorm[k] = t;
    }
}

// ---------------------------------------------------------------------------
// Kernel 2: 128x128 block tile; 8 waves as 4 wave-rows x 2 wave-cols.
// Each wave: 32 rows x 64 cols -> 8 acc tiles, 24 WMMA per 32-wide d-chunk.
// ---------------------------------------------------------------------------
__global__ __launch_bounds__(256) void kq_main_kernel(const float* __restrict__ feats,
                                                      const float* __restrict__ cents,
                                                      const float* __restrict__ cnorm,
                                                      const unsigned short* __restrict__ Chi,
                                                      const unsigned short* __restrict__ Clo,
                                                      int*   __restrict__ labels,
                                                      float* __restrict__ assigned) {
    __shared__ __align__(16) unsigned short Ahi[MTILE * LSTR], Alo[MTILE * LSTR];
    __shared__ __align__(16) unsigned short Bhi[NTILE * LSTR], Blo[NTILE * LSTR];
    __shared__ float cnorm_s[KCNT];
    __shared__ float redv[MTILE][2];
    __shared__ int   redi[MTILE][2];
    __shared__ int   lab_s[MTILE];

    const int tid  = threadIdx.x;
    const int lane = tid & 31;
    const int wave = tid >> 5;      // 0..7
    const int wrow = wave >> 1;     // 0..3 -> 32-row strip
    const int wcol = wave & 1;      // 0..1 -> 64-col strip
    const int half = lane >> 4;     // 0/1
    const int l15  = lane & 15;
    const int row0 = blockIdx.x * MTILE;

    // stage centroid norms (1024 f32)
    {
        float4 v = ((const float4*)cnorm)[tid];
        *(float4*)&cnorm_s[tid * 4] = v;
    }

    float bestv[16];
    int   besti[16];
#pragma unroll
    for (int r = 0; r < 16; ++r) { bestv[r] = 3.4e38f; besti[r] = 0; }

    // staging maps
    const int am_r = tid >> 1;              // A/B staging row 0..127
    const int a_cb = (tid & 1) * 16;        // A: 16-col sub-strip
    const unsigned short* bsrc = (tid & 1) ? Clo : Chi;
    unsigned short*       bdst = (tid & 1) ? Blo : Bhi;

    for (int jb = 0; jb < KCNT; jb += NTILE) {
        v8f acc[8] = {};                    // [s*4+t]: row-strip s (0/1), col-tile t (0..3)

        for (int db = 0; db < DDIM; db += 32) {
            __syncthreads();

            // ---- stage B: 128 centroid rows x 32 dims (bf16 hi/lo), async to LDS
#pragma unroll
            for (int q = 0; q < 4; ++q) {
                async_copy_b128(&bsrc[(size_t)(jb + am_r) * DDIM + db + q * 8],
                                &bdst[am_r * LSTR + q * 8]);
            }
            // ---- stage A: 128 rows x 32 dims f32 -> bf16 hi/lo (VALU overlaps async)
#pragma unroll
            for (int q = 0; q < 4; ++q) {
                const int c = a_cb + q * 4;
                float4 v = *(const float4*)&feats[(size_t)(row0 + am_r) * DDIM + db + c];
                cvt_store4(&Ahi[am_r * LSTR + c], &Alo[am_r * LSTR + c], v);
            }
            if (db + 32 < DDIM) {           // prefetch next feats chunk (global_prefetch_b8)
                __builtin_prefetch(&feats[(size_t)(row0 + am_r) * DDIM + db + 32 + a_cb], 0, 0);
            }
            async_wait0();
            __syncthreads();

            // ---- fragments: two 16-row A strips (hi/lo kept live)
            const int am0 = wrow * 32 + l15;
            const v16bf a0h = load_fragA(&Ahi[am0 * LSTR], half);
            const v16bf a0l = load_fragA(&Alo[am0 * LSTR], half);
            const v16bf a1h = load_fragA(&Ahi[(am0 + 16) * LSTR], half);
            const v16bf a1l = load_fragA(&Alo[(am0 + 16) * LSTR], half);

#pragma unroll
            for (int t = 0; t < 4; ++t) {
                const int bn = wcol * 64 + t * 16 + l15;
                const v16bf bh = load_fragB(&Bhi[bn * LSTR], half);
                const v16bf bl = load_fragB(&Blo[bn * LSTR], half);
                // bf16x3 per tile
                acc[t] = __builtin_amdgcn_wmma_f32_16x16x32_bf16(false, a0h, false, bh, (short)0, acc[t], false, false);
                acc[t] = __builtin_amdgcn_wmma_f32_16x16x32_bf16(false, a0h, false, bl, (short)0, acc[t], false, false);
                acc[t] = __builtin_amdgcn_wmma_f32_16x16x32_bf16(false, a0l, false, bh, (short)0, acc[t], false, false);
                acc[4 + t] = __builtin_amdgcn_wmma_f32_16x16x32_bf16(false, a1h, false, bh, (short)0, acc[4 + t], false, false);
                acc[4 + t] = __builtin_amdgcn_wmma_f32_16x16x32_bf16(false, a1h, false, bl, (short)0, acc[4 + t], false, false);
                acc[4 + t] = __builtin_amdgcn_wmma_f32_16x16x32_bf16(false, a1l, false, bh, (short)0, acc[4 + t], false, false);
            }
        }

        // scores: c_norm - 2*dot ; ascending n with strict < keeps first minimum
#pragma unroll
        for (int t = 0; t < 4; ++t) {
            const int n = jb + wcol * 64 + t * 16 + l15;
            const float cn = cnorm_s[n];
#pragma unroll
            for (int s = 0; s < 2; ++s) {
#pragma unroll
                for (int r = 0; r < 8; ++r) {
                    const float sc = cn - 2.0f * acc[s * 4 + t][r];
                    const int bi = s * 8 + r;
                    if (sc < bestv[bi]) { bestv[bi] = sc; besti[bi] = n; }
                }
            }
        }
    }

    // reduce across the 16 lanes of each half (they share row m)
#pragma unroll
    for (int s = 0; s < 2; ++s) {
#pragma unroll
        for (int r = 0; r < 8; ++r) {
            const int bi = s * 8 + r;
            float v = bestv[bi];
            int   i = besti[bi];
#pragma unroll
            for (int off = 8; off >= 1; off >>= 1) {
                const float ov = __shfl_xor(v, off, 32);
                const int   oi = __shfl_xor(i, off, 32);
                if (ov < v || (ov == v && oi < i)) { v = ov; i = oi; }
            }
            if (l15 == 0) {
                const int m = wrow * 32 + s * 16 + half * 8 + r;
                redv[m][wcol] = v;
                redi[m][wcol] = i;
            }
        }
    }
    __syncthreads();

    // combine the 2 column-strip waves per row; write labels
    if (tid < MTILE) {
        float v = redv[tid][0];
        int   i = redi[tid][0];
        const float ov = redv[tid][1];
        const int   oi = redi[tid][1];
        if (ov < v || (ov == v && oi < i)) { v = ov; i = oi; }
        lab_s[tid] = i;
        labels[row0 + tid] = i;
    }
    __syncthreads();

    // gather assigned centroids: 128 rows x 1024 f32 (one float4 per thread per row)
    for (int r = 0; r < MTILE; ++r) {
        const int lab = lab_s[r];
        float4 v = *(const float4*)&cents[(size_t)lab * DDIM + tid * 4];
        *(float4*)&assigned[(size_t)(row0 + r) * DDIM + tid * 4] = v;
    }
}

// ---------------------------------------------------------------------------
extern "C" void kernel_launch(void* const* d_in, const int* in_sizes, int n_in,
                              void* d_out, int out_size, void* d_ws, size_t ws_size,
                              hipStream_t stream) {
    const float* feats = (const float*)d_in[0];
    const float* cents = (const float*)d_in[1];
    const int N = in_sizes[0] / DDIM;     // 32000
    const int K = in_sizes[1] / DDIM;     // 1024

    // workspace: cnorm f32[K] | Chi bf16[K*D] | Clo bf16[K*D]  (~4.2 MB)
    char* ws = (char*)d_ws;
    float*          cnorm = (float*)ws;
    unsigned short* Chi   = (unsigned short*)(ws + 4096);
    unsigned short* Clo   = (unsigned short*)(ws + 4096 + (size_t)KCNT * DDIM * 2);

    int*   labels   = (int*)d_out;        // int32 labels, N elements
    float* assigned = (float*)d_out + N;  // f32 gathered centroids, N*D elements

    kq_prep_kernel<<<K, 256, 0, stream>>>(cents, cnorm, Chi, Clo);
    kq_main_kernel<<<N / MTILE, 256, 0, stream>>>(feats, cents, cnorm, Chi, Clo,
                                                  labels, assigned);
}
